// MetaLearningWithMemory_80882824118740
// MI455X (gfx1250) — compile-verified, matrix operand order checked by reading, not run on hardware
//
#include <hip/hip_runtime.h>
#include <hip/hip_bf16.h>

typedef __attribute__((ext_vector_type(16))) __bf16 v16bf;
typedef __attribute__((ext_vector_type(8)))  float  v8f;

// ---------------- helpers ----------------

__device__ __forceinline__ unsigned short f32_to_bf16(float f) {
    union { float f; unsigned u; } v; v.f = f;
    unsigned r = v.u + 0x7FFFu + ((v.u >> 16) & 1u);   // round-to-nearest-even
    return (unsigned short)(r >> 16);
}

__device__ __forceinline__ float bf16_to_f32(unsigned short h) {
    union { unsigned u; float f; } v; v.u = ((unsigned)h) << 16;
    return v.f;
}

__device__ __forceinline__ v8f v8f_zero() {
    v8f z;
#pragma unroll
    for (int i = 0; i < 8; ++i) z[i] = 0.f;
    return z;
}

__device__ __forceinline__ v8f wmma_bf16(v16bf a, v16bf b, v8f c) {
    return __builtin_amdgcn_wmma_f32_16x16x32_bf16(
        false, a, false, b, (short)0, c, false, false);
}

// CDNA5 async copy: HBM -> LDS without VGPR staging, tracked by ASYNCcnt.
// VDST VGPR carries the per-lane LDS byte offset (HW adds LDS_BASE); the low
// 32 bits of a generic shared-memory pointer are exactly that offset.
__device__ __forceinline__ void async_b128(const void* gptr, void* lptr) {
    unsigned lds = (unsigned)(size_t)lptr;
    unsigned long long ga = (unsigned long long)(size_t)gptr;
    asm volatile("global_load_async_to_lds_b128 %0, %1, off"
                 :: "v"(lds), "v"(ga) : "memory");
}
__device__ __forceinline__ void wait_async0() {
    asm volatile("s_wait_asynccnt 0" ::: "memory");
}

// 16x32 bf16 fragment from LDS [row][k] (rowStride in ushorts, even).
// CDNA5 16-bit A-matrix layout: lane l -> row (l&15); lane-half picks K-halves;
// VGPR v: v<4 -> K=2v+8*half ; v>=4 -> K=16+2(v-4)+8*half.  The two 4-dword
// halves are k-contiguous, so the compiler merges them into ds_load_b128 pairs.
__device__ __forceinline__ v16bf load_frag(const unsigned short* base, int rowStride,
                                           int row, int kbase, int lane) {
    const int rr = row + (lane & 15);
    const int khalf = lane >> 4;
    const unsigned short* p = base + (size_t)rr * rowStride + kbase;
    union { v16bf v; unsigned u[8]; } f;
#pragma unroll
    for (int v = 0; v < 8; ++v) {
        int kk = (v < 4) ? (2 * v + 8 * khalf) : (16 + 2 * (v - 4) + 8 * khalf);
        f.u[v] = *(const unsigned*)(p + kk);
    }
    return f.v;
}

// ---------------- conversion kernels (run once per operand) ----------------

__global__ __launch_bounds__(256) void convert_bf16_kernel(
    const float* __restrict__ src, unsigned short* __restrict__ dst, size_t n8)
{
    size_t i = (size_t)blockIdx.x * 256 + threadIdx.x;
    if (i >= n8) return;
    const float4 a = *(const float4*)(src + i * 8);
    const float4 b = *(const float4*)(src + i * 8 + 4);
    uint4 o;
    o.x = (unsigned)f32_to_bf16(a.x) | ((unsigned)f32_to_bf16(a.y) << 16);
    o.y = (unsigned)f32_to_bf16(a.z) | ((unsigned)f32_to_bf16(a.w) << 16);
    o.z = (unsigned)f32_to_bf16(b.x) | ((unsigned)f32_to_bf16(b.y) << 16);
    o.w = (unsigned)f32_to_bf16(b.z) | ((unsigned)f32_to_bf16(b.w) << 16);
    *(uint4*)(dst + i * 8) = o;
}

// f32 [R,C] -> bf16 [C,R] via 32x32 LDS tile (R,C multiples of 32)
__global__ __launch_bounds__(256) void transpose_bf16_kernel(
    const float* __restrict__ src, unsigned short* __restrict__ dst, int R, int C)
{
    __shared__ unsigned short tile[32][33];
    const int bx = blockIdx.x * 32;   // C
    const int by = blockIdx.y * 32;   // R
    const int tx = threadIdx.x & 31, ty = threadIdx.x >> 5;
#pragma unroll
    for (int i = 0; i < 32; i += 8)
        tile[ty + i][tx] = f32_to_bf16(src[(size_t)(by + ty + i) * C + bx + tx]);
    __syncthreads();
#pragma unroll
    for (int i = 0; i < 32; i += 8)
        dst[(size_t)(bx + ty + i) * R + by + tx] = tile[tx][ty + i];
}

// ---------------- bf16-WMMA GEMM, async-staged, double-buffered ----------------
// C = A[M,K]bf16 @ Bt[N,K]bf16^T + bias (+Cadd f32). Outputs f32 and/or bf16.
// 128x128x32 tiles, 256 threads = 8 waves (2M x 4N), wave tile 64x32 = 4x2 WMMA.
// Tile t+1 streams HBM->LDS via global_load_async_to_lds_b128 (no VGPRs, no
// spills) while tile t's WMMAs run; s_wait_asynccnt 0 + barrier per k-step.
// Requires M%128==0, N%128==0, K%32==0.

#define LSH 40   // LDS row stride in ushorts (80B: 16B-aligned, conflict-light)

__global__ __launch_bounds__(256) void gemm_bf16_kernel(
    const unsigned short* __restrict__ A, int lda,
    const unsigned short* __restrict__ Bt, int ldb,
    const float* __restrict__ bias,
    const float* __restrict__ Cadd, int ldcadd,
    float* __restrict__ Cf, unsigned short* __restrict__ Cbf,
    int ldc, int K)
{
    __shared__ __align__(16) unsigned short Ash[2][128 * LSH];
    __shared__ __align__(16) unsigned short Bsh[2][128 * LSH];

    const int tid  = threadIdx.x;
    const int lane = tid & 31;
    const int wid  = tid >> 5;
    const int wm   = wid & 1;
    const int wn   = wid >> 1;
    const int m0   = blockIdx.y * 128;
    const int n0   = blockIdx.x * 128;

    // each thread moves 2 x 16B chunks of A and of B per 128x32 tile
    const int srow = tid >> 2;            // rows 0..63 (+64 for r=1)
    const int scol = (tid & 3) * 8;       // ushort offset within 32-wide k

    auto stage_async = [&](int buf, int k0) {
#pragma unroll
        for (int r = 0; r < 2; ++r) {
            int row = srow + r * 64;
            async_b128(A  + (size_t)(m0 + row) * lda + k0 + scol,
                       &Ash[buf][row * LSH + scol]);
            async_b128(Bt + (size_t)(n0 + row) * ldb + k0 + scol,
                       &Bsh[buf][row * LSH + scol]);
        }
    };

    v8f acc[4][2];
#pragma unroll
    for (int i = 0; i < 4; ++i)
#pragma unroll
        for (int j = 0; j < 2; ++j) acc[i][j] = v8f_zero();

    stage_async(0, 0);
    wait_async0();
    __syncthreads();

    int buf = 0;
    for (int k0 = 0; k0 < K; k0 += 32) {
        const bool more = (k0 + 32) < K;
        if (more) stage_async(buf ^ 1, k0 + 32);       // overlaps WMMAs below
        if (k0 + 64 < K) {                             // gfx1250 global_prefetch_b8
            __builtin_prefetch((const void*)(A  + (size_t)(m0 + srow) * lda + k0 + 64 + scol), 0, 3);
            __builtin_prefetch((const void*)(Bt + (size_t)(n0 + srow) * ldb + k0 + 64 + scol), 0, 3);
        }

        v16bf afrag[4], bfrag[2];
#pragma unroll
        for (int tm = 0; tm < 4; ++tm)
            afrag[tm] = load_frag(Ash[buf], LSH, wm * 64 + tm * 16, 0, lane);
#pragma unroll
        for (int tn = 0; tn < 2; ++tn)
            bfrag[tn] = load_frag(Bsh[buf], LSH, wn * 32 + tn * 16, 0, lane);
#pragma unroll
        for (int tm = 0; tm < 4; ++tm)
#pragma unroll
            for (int tn = 0; tn < 2; ++tn)
                acc[tm][tn] = wmma_bf16(afrag[tm], bfrag[tn], acc[tm][tn]);

        if (more) wait_async0();
        __syncthreads();
        buf ^= 1;
    }

    const int lh = lane >> 4;
    const int ln = lane & 15;
#pragma unroll
    for (int tm = 0; tm < 4; ++tm) {
#pragma unroll
        for (int tn = 0; tn < 2; ++tn) {
            int ng = n0 + wn * 32 + tn * 16 + ln;
            float bv = bias ? bias[ng] : 0.f;
#pragma unroll
            for (int r = 0; r < 8; ++r) {
                int mg = m0 + wm * 64 + tm * 16 + r + 8 * lh;
                float v = acc[tm][tn][r] + bv;
                if (Cadd) v += Cadd[(size_t)mg * ldcadd + ng];
                if (Cf)  Cf [(size_t)mg * ldc + ng] = v;
                if (Cbf) Cbf[(size_t)mg * ldc + ng] = f32_to_bf16(v);
            }
        }
    }
}

// ---------------- sequential memory-write scan ----------------
// scores_i = S0[:,i] + W[:, :i] @ G[:i, i]  (G symmetric); w_i = softmax(./sqrt F).
// History W lives in LDS as bf16 (256 KB fits the 320 KB WGP LDS). 1 WG, 256 thr.

__global__ __launch_bounds__(256) void scan_kernel(
    const float* __restrict__ G,      // [512,512]
    const float* __restrict__ S0,     // [256,512]
    const int*   __restrict__ support_y,
    const float* __restrict__ mem_values,
    unsigned short* __restrict__ Wbf, // out [256,512] bf16
    float* __restrict__ mv)           // out [256,512] f32
{
    extern __shared__ unsigned char smem[];
    unsigned short* Wl  = (unsigned short*)smem;            // 512*256 bf16
    float* Grow = (float*)(smem + 512 * 256 * 2);           // 512
    float* red  = Grow + 512;                               // 256

    const int m = threadIdx.x;
    const float inv_sqrt_f = 0.04419417382415922f;          // 512^-0.5
    float acc5[5] = {0.f, 0.f, 0.f, 0.f, 0.f};

    for (int i = 0; i < 512; ++i) {
        __syncthreads();
        Grow[m]       = G[(size_t)i * 512 + m];
        Grow[m + 256] = G[(size_t)i * 512 + m + 256];
        __syncthreads();

        float s = S0[(size_t)m * 512 + i];
        int j = 0;
        for (; j + 4 <= i; j += 4) {
            s += bf16_to_f32(Wl[(j + 0) * 256 + m]) * Grow[j + 0];
            s += bf16_to_f32(Wl[(j + 1) * 256 + m]) * Grow[j + 1];
            s += bf16_to_f32(Wl[(j + 2) * 256 + m]) * Grow[j + 2];
            s += bf16_to_f32(Wl[(j + 3) * 256 + m]) * Grow[j + 3];
        }
        for (; j < i; ++j) s += bf16_to_f32(Wl[j * 256 + m]) * Grow[j];
        s *= inv_sqrt_f;

        red[m] = s;
        __syncthreads();
        for (int st = 128; st > 0; st >>= 1) {
            if (m < st) red[m] = fmaxf(red[m], red[m + st]);
            __syncthreads();
        }
        float mx = red[0];
        __syncthreads();
        float e = __expf(s - mx);
        red[m] = e;
        __syncthreads();
        for (int st = 128; st > 0; st >>= 1) {
            if (m < st) red[m] = red[m] + red[m + st];
            __syncthreads();
        }
        float w = e / red[0];

        unsigned short wb = f32_to_bf16(w);
        Wl[i * 256 + m] = wb;
        Wbf[(size_t)m * 512 + i] = wb;
        int y = support_y[i];
#pragma unroll
        for (int c = 0; c < 5; ++c)
            if (c == y) acc5[c] += w;
    }

    for (int d = 0; d < 512; ++d) {
        float v = mem_values[(size_t)m * 512 + d];
        if (d < 5) v += acc5[d];
        mv[(size_t)m * 512 + d] = v;
    }
}

// ---------------- fused multi-head attention over the memory bank ----------------
// Block = 128 query rows, 256 threads (8 waves x 16 rows). All operands arrive
// pre-converted bf16 (V pre-transposed); staging is async HBM->LDS b128 copies.

__global__ __launch_bounds__(256) void attention_kernel(
    const unsigned short* __restrict__ qfbf,  // [16384,512] bf16
    const unsigned short* __restrict__ mkbf,  // [256,512]  bf16
    const unsigned short* __restrict__ mvT,   // [512,256]  bf16 (mv transposed)
    float* __restrict__ mem_out)              // [16384,512] f32
{
    extern __shared__ unsigned char smem[];
    unsigned short* Ksh = (unsigned short*)smem;   // [256][72]
    unsigned short* Vsh = Ksh + 256 * 72;          // [64][264]
    unsigned short* Qsh = Vsh + 64 * 264;          // [128][72]
    unsigned short* Psh = Qsh + 128 * 72;          // [128][264]

    const int tid  = threadIdx.x;
    const int lane = tid & 31;
    const int wid  = tid >> 5;
    const int lh   = lane >> 4;
    const int ln   = lane & 15;
    const int bRow = blockIdx.x * 128;

    for (int h = 0; h < 8; ++h) {
        __syncthreads();                       // prior head done with K/V/Q LDS
        for (int idx = tid; idx < 2048; idx += 256) {          // K_h: 256x64
            int memi = idx >> 3, c8 = (idx & 7) * 8;
            async_b128(mkbf + (size_t)memi * 512 + h * 64 + c8,
                       &Ksh[memi * 72 + c8]);
        }
        for (int idx = tid; idx < 2048; idx += 256) {          // V_h^T: 64x256
            int d = idx >> 5, c8 = (idx & 31) * 8;
            async_b128(mvT + (size_t)(h * 64 + d) * 256 + c8,
                       &Vsh[d * 264 + c8]);
        }
        for (int idx = tid; idx < 1024; idx += 256) {          // Q: 128x64
            int r = idx >> 3, c8 = (idx & 7) * 8;
            async_b128(qfbf + (size_t)(bRow + r) * 512 + h * 64 + c8,
                       &Qsh[r * 72 + c8]);
        }
        wait_async0();
        __syncthreads();

        // scores [16 x 256] = Q @ K_h^T  (K=64 -> 2 WMMA k-steps)
        v8f sacc[16];
#pragma unroll
        for (int t = 0; t < 16; ++t) sacc[t] = v8f_zero();
#pragma unroll
        for (int ks = 0; ks < 2; ++ks) {
            v16bf a = load_frag(Qsh, 72, wid * 16, ks * 32, lane);
#pragma unroll
            for (int t = 0; t < 16; ++t) {
                v16bf b = load_frag(Ksh, 72, t * 16, ks * 32, lane);
                sacc[t] = wmma_bf16(a, b, sacc[t]);
            }
        }

        const float scale = 0.125f;  // 64^-0.5
        float rowsum[8];
#pragma unroll
        for (int r = 0; r < 8; ++r) {
            float mx = -1e30f;
#pragma unroll
            for (int t = 0; t < 16; ++t) mx = fmaxf(mx, sacc[t][r]);
            mx = fmaxf(mx, __shfl_xor(mx, 1));
            mx = fmaxf(mx, __shfl_xor(mx, 2));
            mx = fmaxf(mx, __shfl_xor(mx, 4));
            mx = fmaxf(mx, __shfl_xor(mx, 8));
            float ssum = 0.f;
            int prow = wid * 16 + r + 8 * lh;
#pragma unroll
            for (int t = 0; t < 16; ++t) {
                float p = __expf((sacc[t][r] - mx) * scale);
                ssum += p;
                Psh[prow * 264 + t * 16 + ln] = f32_to_bf16(p);
            }
            ssum += __shfl_xor(ssum, 1);
            ssum += __shfl_xor(ssum, 2);
            ssum += __shfl_xor(ssum, 4);
            ssum += __shfl_xor(ssum, 8);
            rowsum[r] = ssum;
        }
        // same-wave LDS producer->consumer: per-wave DScnt ordering suffices

        // O [16 x 64] = P @ V_h  (K=256 -> 8 WMMA k-steps)
        v8f oacc[4];
#pragma unroll
        for (int tn = 0; tn < 4; ++tn) oacc[tn] = v8f_zero();
#pragma unroll
        for (int ks = 0; ks < 8; ++ks) {
            v16bf a = load_frag(Psh, 264, wid * 16, ks * 32, lane);
#pragma unroll
            for (int tn = 0; tn < 4; ++tn) {
                v16bf b = load_frag(Vsh, 264, tn * 16, ks * 32, lane);
                oacc[tn] = wmma_bf16(a, b, oacc[tn]);
            }
        }
#pragma unroll
        for (int tn = 0; tn < 4; ++tn) {
#pragma unroll
            for (int r = 0; r < 8; ++r) {
                int mg = bRow + wid * 16 + r + 8 * lh;
                int ng = h * 64 + tn * 16 + ln;
                mem_out[(size_t)mg * 512 + ng] = oacc[tn][r] / rowsum[r];
            }
        }
    }
}

// ---------------- classifier: [B,1024] @ [1024,5] + b ----------------
__global__ __launch_bounds__(256) void classifier_kernel(
    const unsigned short* __restrict__ feats,   // bf16 [B,512]
    const float* __restrict__ memout,           // f32  [B,512]
    const float* __restrict__ Wcls, const float* __restrict__ bcls,
    float* __restrict__ out, int Bn)
{
    int b = blockIdx.x * blockDim.x + threadIdx.x;
    if (b >= Bn) return;
    float acc[5];
#pragma unroll
    for (int c = 0; c < 5; ++c) acc[c] = bcls[c];
    for (int f = 0; f < 512; ++f) {
        float v = bf16_to_f32(feats[(size_t)b * 512 + f]);
#pragma unroll
        for (int c = 0; c < 5; ++c) acc[c] += v * Wcls[f * 5 + c];
    }
    for (int f = 0; f < 512; ++f) {
        float v = memout[(size_t)b * 512 + f];
#pragma unroll
        for (int c = 0; c < 5; ++c) acc[c] += v * Wcls[(512 + f) * 5 + c];
    }
#pragma unroll
    for (int c = 0; c < 5; ++c) out[(size_t)b * 5 + c] = acc[c];
}

// ---------------- launch ----------------

extern "C" void kernel_launch(void* const* d_in, const int* in_sizes, int n_in,
                              void* d_out, int out_size, void* d_ws, size_t ws_size,
                              hipStream_t stream) {
    (void)in_sizes; (void)n_in; (void)out_size; (void)ws_size;

    const float* x          = (const float*)d_in[0];
    const float* support_x  = (const float*)d_in[1];
    const int*   support_y  = (const int*)  d_in[2];
    const float* W_enc      = (const float*)d_in[3];
    const float* b_enc      = (const float*)d_in[4];
    const float* W_q        = (const float*)d_in[5];
    const float* b_q        = (const float*)d_in[6];
    const float* W_cls      = (const float*)d_in[7];
    const float* b_cls      = (const float*)d_in[8];
    const float* mem_keys   = (const float*)d_in[9];
    const float* mem_values = (const float*)d_in[10];
    float*       out        = (float*)d_out;

    // -------- workspace carve (256B aligned chunks), ~110 MB --------
    size_t off = 0;
    auto alloc_us = [&](size_t e) { unsigned short* p = (unsigned short*)((char*)d_ws + off);
                                    off += ((e * 2 + 255) & ~(size_t)255); return p; };
    auto alloc_f  = [&](size_t e) { float* p = (float*)((char*)d_ws + off);
                                    off += ((e * 4 + 255) & ~(size_t)255); return p; };

    unsigned short* xbf     = alloc_us((size_t)16384 * 2048);
    float*          mem_out = (float*)xbf;          // alias: xbf dead after GEMM 1
    unsigned short* sxbf    = alloc_us((size_t)512 * 2048);
    unsigned short* Wenc_t  = alloc_us((size_t)512 * 2048);
    unsigned short* Wq_t    = alloc_us((size_t)512 * 512);
    unsigned short* featbf  = alloc_us((size_t)16384 * 512);
    unsigned short* qfbf    = alloc_us((size_t)16384 * 512);
    float*          Ksf     = alloc_f ((size_t)512 * 512);
    unsigned short* Ksbf    = alloc_us((size_t)512 * 512);
    unsigned short* Kstbf   = alloc_us((size_t)512 * 512);
    unsigned short* mkeysbf = alloc_us((size_t)256 * 512);
    float*          Gm      = alloc_f ((size_t)512 * 512);
    float*          S0      = alloc_f ((size_t)256 * 512);
    unsigned short* Wbf     = alloc_us((size_t)256 * 512);
    float*          mv      = alloc_f ((size_t)256 * 512);
    unsigned short* mvT     = alloc_us((size_t)512 * 256);
    unsigned short* mkbf    = alloc_us((size_t)256 * 512);

    const size_t scanLds = 512 * 256 * 2 + (512 + 256) * sizeof(float);
    const size_t attnLds = (size_t)(256 * 72 + 64 * 264 + 128 * 72 + 128 * 264) * 2;
    hipFuncSetAttribute(reinterpret_cast<const void*>(scan_kernel),
                        hipFuncAttributeMaxDynamicSharedMemorySize, (int)scanLds);
    hipFuncSetAttribute(reinterpret_cast<const void*>(attention_kernel),
                        hipFuncAttributeMaxDynamicSharedMemorySize, (int)attnLds);

    dim3 blk(256);

    // one-time precision/layout conversions
    convert_bf16_kernel<<<16384, blk, 0, stream>>>(x, xbf, (size_t)16384 * 2048 / 8);
    convert_bf16_kernel<<<512,   blk, 0, stream>>>(support_x, sxbf, (size_t)512 * 2048 / 8);
    convert_bf16_kernel<<<64,    blk, 0, stream>>>(mem_keys, mkeysbf, (size_t)256 * 512 / 8);
    transpose_bf16_kernel<<<dim3(16, 64), blk, 0, stream>>>(W_enc, Wenc_t, 2048, 512);
    transpose_bf16_kernel<<<dim3(16, 16), blk, 0, stream>>>(W_q, Wq_t, 512, 512);

    // 1) featbf = x @ W_enc + b_enc            (M=16384, N=512, K=2048)
    gemm_bf16_kernel<<<dim3(4, 128), blk, 0, stream>>>(
        xbf, 2048, Wenc_t, 2048, b_enc, nullptr, 0, nullptr, featbf, 512, 2048);
    // 2) Ks(f32+bf16) = support_x @ W_enc + b_enc
    gemm_bf16_kernel<<<dim3(4, 4), blk, 0, stream>>>(
        sxbf, 2048, Wenc_t, 2048, b_enc, nullptr, 0, Ksf, Ksbf, 512, 2048);
    transpose_bf16_kernel<<<dim3(16, 16), blk, 0, stream>>>(Ksf, Kstbf, 512, 512);
    // 3) G = Ks @ Ks^T   (Bt = Ks itself)
    gemm_bf16_kernel<<<dim3(4, 4), blk, 0, stream>>>(
        Ksbf, 512, Ksbf, 512, nullptr, nullptr, 0, Gm, nullptr, 512, 512);
    // 4) S0 = mem_keys @ Ks^T
    gemm_bf16_kernel<<<dim3(4, 2), blk, 0, stream>>>(
        mkeysbf, 512, Ksbf, 512, nullptr, nullptr, 0, S0, nullptr, 512, 512);
    // 5) sequential scan -> W (bf16), mv (f32)
    scan_kernel<<<1, 256, scanLds, stream>>>(Gm, S0, support_y, mem_values, Wbf, mv);
    transpose_bf16_kernel<<<dim3(16, 8), blk, 0, stream>>>(mv, mvT, 256, 512);
    // 6) mk = W @ Ks + mem_keys   (B = Ks -> Bt = Ks^T = Kstbf)
    gemm_bf16_kernel<<<dim3(4, 2), blk, 0, stream>>>(
        Wbf, 512, Kstbf, 512, nullptr, mem_keys, 512, nullptr, mkbf, 512, 512);
    // 7) qf = features @ W_q + b_q
    gemm_bf16_kernel<<<dim3(4, 128), blk, 0, stream>>>(
        featbf, 512, Wq_t, 512, b_q, nullptr, 0, nullptr, qfbf, 512, 512);
    // 8) fused MHA over the memory bank
    attention_kernel<<<128, blk, attnLds, stream>>>(qfbf, mkbf, mvT, mem_out);
    // 9) logits = [features, mem_out] @ W_cls + b_cls
    classifier_kernel<<<64, blk, 0, stream>>>(featbf, mem_out, W_cls, b_cls, out, 16384);
}
